// CrossLevelAttention_45732811768403
// MI455X (gfx1250) — compile-verified, matrix operand order checked by reading, not run on hardware
//
#include <hip/hip_runtime.h>
#include <hip/hip_bf16.h>
#include <math.h>

// ---------------------------------------------------------------------------
// CDNA5 (gfx1250) wave32 WMMA implementation of CrossLevelAttention forward.
// Big GEMMs: v_wmma_f32_16x16x32_bf16; fragments are 2x ds_load_b128; each
// barrier pair feeds 4 WMMAs (K-step 64, two N-tiles per wave, shared A
// fragment).  Score-row load in pooling uses the Tensor Data Mover.
// ---------------------------------------------------------------------------

typedef __attribute__((ext_vector_type(16))) __bf16 bf16x16;
typedef __attribute__((ext_vector_type(8)))  __bf16 bf16x8;
typedef __attribute__((ext_vector_type(8)))  float  f32x8;
typedef __attribute__((ext_vector_type(4)))  unsigned u32x4;
typedef __attribute__((ext_vector_type(8)))  int      i32x8;
typedef __attribute__((ext_vector_type(4)))  int      i32x4;

#define BATCH 32
#define COMMON 1024
#define TIME_DIM 256

#if defined(__has_builtin)
#if __has_builtin(__builtin_amdgcn_tensor_load_to_lds) && __has_builtin(__builtin_amdgcn_s_wait_tensorcnt)
#define HAVE_TDM 1
#endif
#endif

__device__ __forceinline__ unsigned bf16_bits(float f) {
  unsigned u = __builtin_bit_cast(unsigned, f);
  return (u + 0x7FFFu + ((u >> 16) & 1u)) >> 16;   // round-to-nearest-even
}
__device__ __forceinline__ unsigned pack2_bf16(float lo, float hi) {
  return bf16_bits(lo) | (bf16_bits(hi) << 16);
}

__device__ __forceinline__ float gelu_tanh(float x) {
  return 0.5f * x * (1.0f + tanhf(0.7978845608028654f * (x + 0.044715f * x * x * x)));
}

// Two contiguous 8-element bf16 runs -> one 16-element fragment (2x b128).
__device__ __forceinline__ bf16x16 frag2(const __bf16* p0, const __bf16* p1) {
  bf16x8 lo = *reinterpret_cast<const bf16x8*>(p0);
  bf16x8 hi = *reinterpret_cast<const bf16x8*>(p1);
  return __builtin_shufflevector(lo, hi, 0, 1, 2, 3, 4, 5, 6, 7,
                                 8, 9, 10, 11, 12, 13, 14, 15);
}

// Wave32 16-bit fragment layout: r = lane&15, half = lane>>4; each lane holds
// K runs {half*8..+7} and {16+half*8..+7}.
__device__ __forceinline__ bf16x16 load_fragA(const __bf16* lA, int ldA, int k0,
                                              int r, int half) {
  const __bf16* pa = lA + r * ldA + k0 + half * 8;
  return frag2(pa, pa + 16);
}
// lBT layout: [128 cols][64 k] bf16 (lBT[c*64 + k]); per-column K runs are
// contiguous so B fragments are also 2x b128.
__device__ __forceinline__ bf16x16 load_fragB(const __bf16* lBT, int col, int ksub,
                                              int half) {
  const __bf16* pb = lBT + col * 64 + ksub + half * 8;
  return frag2(pb, pb + 16);
}

__device__ __forceinline__ f32x8 wmma_bf16(bf16x16 a, bf16x16 b, f32x8 acc) {
  return __builtin_amdgcn_wmma_f32_16x16x32_bf16(false, a, false, b,
                                                 (short)0, acc, false, false);
}

// Stage A tile (rows x cols fp32, row-major, cols multiple of 4) -> bf16 LDS.
__device__ __forceinline__ void stage_A_bf16(__bf16* dst, const float* __restrict__ src,
                                             int rows, int cols, int ldsrc,
                                             int tid, int nthreads) {
  const int total = rows * cols;
  for (int idx = tid * 4; idx < total; idx += nthreads * 4) {
    const int rr = idx / cols, cc = idx % cols;
    const float4 f = *reinterpret_cast<const float4*>(src + (size_t)rr * ldsrc + cc);
    unsigned* d = reinterpret_cast<unsigned*>(dst + idx);
    d[0] = pack2_bf16(f.x, f.y);
    d[1] = pack2_bf16(f.z, f.w);
  }
}

// Stage B K-slice (64 k x 128 c fp32 at src[k*ldsrc + c]) transposed into
// lBT[c*64 + k] bf16.  Global reads coalesced along c; LDS writes are b64.
__device__ __forceinline__ void stage_Bt_bf16(__bf16* dst, const float* __restrict__ src,
                                              int ldsrc, int tid) {
  for (int q = tid; q < 2048; q += 128) {    // 2048 quads of 4 k-values
    const int c = q & 127;
    const int k = (q >> 7) * 4;
    const float f0 = src[(size_t)(k + 0) * ldsrc + c];
    const float f1 = src[(size_t)(k + 1) * ldsrc + c];
    const float f2 = src[(size_t)(k + 2) * ldsrc + c];
    const float f3 = src[(size_t)(k + 3) * ldsrc + c];
    unsigned* d = reinterpret_cast<unsigned*>(dst + c * 64 + k);
    d[0] = pack2_bf16(f0, f1);
    d[1] = pack2_bf16(f2, f3);
  }
}

// ---------------------------------------------------------------------------
// Tensor Data Mover: 1-D row load (nelem fp32) into LDS.  Issued per-wave.
// ---------------------------------------------------------------------------
__device__ __forceinline__ void tdm_load_row(float* ldst, const float* gsrc, int nelem) {
#ifdef HAVE_TDM
  const unsigned long long ga = (unsigned long long)(const void*)gsrc;
  const unsigned lds_off = (unsigned)(unsigned long long)(void*)ldst;
  u32x4 g0 = { 1u,                                   // count=1
               lds_off,                              // lds_addr
               (unsigned)ga,                         // global_addr[31:0]
               (unsigned)((ga >> 32) & 0x1FFFFFFull) | 0x80000000u }; // addr[56:32] | type=2
  i32x8 g1 = { (int)0x20000u,                        // data_size=4B
               (int)(((unsigned)nelem & 0xFFFFu) << 16),                // tensor_dim0[15:0]
               (int)((((unsigned)nelem >> 16) & 0xFFFFu) | (1u << 16)), // dim0 hi, dim1=1
               (int)((unsigned)nelem << 16),          // tile_dim0
               1,                                     // tile_dim1=1
               nelem,                                 // tensor_dim0_stride
               0, 0 };
  i32x4 z4 = {0, 0, 0, 0};
#if __clang_major__ >= 23
  i32x8 z8 = {0, 0, 0, 0, 0, 0, 0, 0};
  __builtin_amdgcn_tensor_load_to_lds(g0, g1, z4, z4, z8, 0);
#else
  __builtin_amdgcn_tensor_load_to_lds(g0, g1, z4, z4, 0);
#endif
  __builtin_amdgcn_s_wait_tensorcnt(0);
#else
  (void)ldst; (void)gsrc; (void)nelem;
#endif
}

// ---------------------------------------------------------------------------
// Kernel 1: attention-pool scores.  score[b,s] = gelu(e_row @ W1 + b1) . W2
// Block: 128 threads, 16 rows; N processed in 128-col chunks, 4 WMMA/barrier.
// ---------------------------------------------------------------------------
template <int D, int S>
__global__ void score_kernel(const float* __restrict__ e,
                             const float* __restrict__ W1,
                             const float* __restrict__ b1,
                             const float* __restrict__ W2,
                             float* __restrict__ sc) {
  constexpr int N = D / 2;
  __shared__ __bf16 lA[16 * D];
  __shared__ __bf16 lBT[128 * 64];
  __shared__ float  lred[16];

  const int tid = threadIdx.x, lane = tid & 31, wave = tid >> 5;
  const int half = lane >> 4, r = lane & 15;
  const int col0 = wave * 16 + r;           // local cols: col0 and col0+64
  const int row0 = blockIdx.x * 16;

  stage_A_bf16(lA, e + (size_t)row0 * D, 16, D, D, tid, 128);
  if (tid < 16) lred[tid] = 0.0f;
  __syncthreads();

  float racc[8];
#pragma unroll
  for (int v = 0; v < 8; ++v) racc[v] = 0.0f;

  for (int cb = 0; cb < N; cb += 128) {
    f32x8 acc0 = {0, 0, 0, 0, 0, 0, 0, 0};
    f32x8 acc1 = {0, 0, 0, 0, 0, 0, 0, 0};
    for (int k0 = 0; k0 < D; k0 += 64) {
      __syncthreads();
      stage_Bt_bf16(lBT, W1 + (size_t)k0 * N + cb, N, tid);
      if (k0 + 64 < D)
        __builtin_prefetch(W1 + (size_t)(k0 + 64) * N + cb, 0, 1);
      __syncthreads();
#pragma unroll
      for (int ks = 0; ks < 64; ks += 32) {
        bf16x16 a  = load_fragA(lA, D, k0 + ks, r, half);
        bf16x16 b0 = load_fragB(lBT, col0, ks, half);
        bf16x16 b1f = load_fragB(lBT, col0 + 64, ks, half);
        acc0 = wmma_bf16(a, b0, acc0);
        acc1 = wmma_bf16(a, b1f, acc1);
      }
    }
    const int c0 = cb + col0, c1 = cb + col0 + 64;
    const float w20 = W2[c0], b10 = b1[c0];
    const float w21 = W2[c1], b11 = b1[c1];
#pragma unroll
    for (int v = 0; v < 8; ++v) {
      racc[v] += gelu_tanh(acc0[v] + b10) * w20;
      racc[v] += gelu_tanh(acc1[v] + b11) * w21;
    }
  }

#pragma unroll
  for (int v = 0; v < 8; ++v)
    for (int off = 1; off < 16; off <<= 1)
      racc[v] += __shfl_xor(racc[v], off, 16);
  if ((lane & 15) == 0) {
#pragma unroll
    for (int v = 0; v < 8; ++v) atomicAdd(&lred[v + half * 8], racc[v]);
  }
  __syncthreads();
  if (tid < 16) sc[row0 + tid] = lred[tid];   // softmax-invariant bias b2 dropped
}

// ---------------------------------------------------------------------------
// Kernel 2: softmax over S + weighted pooling.  One block per batch.
// Score row is DMA'd into LDS once via the Tensor Data Mover.
// ---------------------------------------------------------------------------
template <int D, int S>
__global__ void pool_kernel(const float* __restrict__ e,
                            const float* __restrict__ sc,
                            float* __restrict__ pooled) {
  const int b = blockIdx.x, tid = threadIdx.x;
  __shared__ float sraw[S];
  __shared__ float w[S];
  __shared__ float red[256];
  const float* scb = sc + (size_t)b * S;

#ifdef HAVE_TDM
  if (tid < 32) tdm_load_row(sraw, scb, S);   // wave 0 issues the TDM copy
#else
  for (int s = tid; s < S; s += 256) sraw[s] = scb[s];
#endif
  __syncthreads();

  float mx = -1e30f;
  for (int s = tid; s < S; s += 256) mx = fmaxf(mx, sraw[s]);
  red[tid] = mx; __syncthreads();
  for (int o = 128; o > 0; o >>= 1) { if (tid < o) red[tid] = fmaxf(red[tid], red[tid + o]); __syncthreads(); }
  mx = red[0]; __syncthreads();

  float sm = 0.0f;
  for (int s = tid; s < S; s += 256) { float v = __expf(sraw[s] - mx); w[s] = v; sm += v; }
  red[tid] = sm; __syncthreads();
  for (int o = 128; o > 0; o >>= 1) { if (tid < o) red[tid] += red[tid + o]; __syncthreads(); }
  const float inv = 1.0f / red[0]; __syncthreads();
  for (int s = tid; s < S; s += 256) w[s] *= inv;
  __syncthreads();

  for (int c = tid; c < D; c += 256) {
    float acc = 0.0f;
    const float* ep = e + (size_t)b * S * D + c;
    for (int s = 0; s < S; ++s) acc += w[s] * ep[(size_t)s * D];
    pooled[(size_t)b * D + c] = acc;
  }
}

// ---------------------------------------------------------------------------
// Kernel 3: level projection + positional embedding -> x[b, lvl, 1024]
// ---------------------------------------------------------------------------
__global__ void proj_kernel(const float* __restrict__ pooled,
                            const float* __restrict__ W,
                            const float* __restrict__ bias,
                            const float* __restrict__ pos,
                            float* __restrict__ x, int D, int lvl) {
  const int b = blockIdx.x;
  __shared__ float p[1024];
  for (int k = threadIdx.x; k < D; k += 256) p[k] = pooled[(size_t)b * D + k];
  __syncthreads();
#pragma unroll
  for (int j = 0; j < 4; ++j) {
    const int c = threadIdx.x + j * 256;
    float acc = bias[c] + pos[lvl * COMMON + c];
    for (int k = 0; k < D; ++k) acc += p[k] * W[(size_t)k * COMMON + c];
    x[((size_t)b * 4 + lvl) * COMMON + c] = acc;
  }
}

// ---------------------------------------------------------------------------
// Kernel 4: WMMA linear for the MHA projections.
// out[128,1024] = A[128,1024] @ W[1024,1024] + bias.  Grid (8, 8), 128 thr.
// ---------------------------------------------------------------------------
__global__ void linear128_kernel(const float* __restrict__ A,
                                 const float* __restrict__ W,
                                 const float* __restrict__ bias,
                                 float* __restrict__ out) {
  __shared__ __bf16 lA[16 * 1024];
  __shared__ __bf16 lBT[128 * 64];
  const int tid = threadIdx.x, lane = tid & 31, wave = tid >> 5;
  const int half = lane >> 4, r = lane & 15;
  const int col0 = wave * 16 + r;
  const int row0 = blockIdx.x * 16, cbase = blockIdx.y * 128;

  stage_A_bf16(lA, A + (size_t)row0 * 1024, 16, 1024, 1024, tid, 128);
  __syncthreads();

  f32x8 acc0 = {0, 0, 0, 0, 0, 0, 0, 0};
  f32x8 acc1 = {0, 0, 0, 0, 0, 0, 0, 0};
  for (int k0 = 0; k0 < 1024; k0 += 64) {
    __syncthreads();
    stage_Bt_bf16(lBT, W + (size_t)k0 * 1024 + cbase, 1024, tid);
    if (k0 + 64 < 1024)
      __builtin_prefetch(W + (size_t)(k0 + 64) * 1024 + cbase, 0, 1);
    __syncthreads();
#pragma unroll
    for (int ks = 0; ks < 64; ks += 32) {
      bf16x16 a  = load_fragA(lA, 1024, k0 + ks, r, half);
      bf16x16 b0 = load_fragB(lBT, col0, ks, half);
      bf16x16 b1f = load_fragB(lBT, col0 + 64, ks, half);
      acc0 = wmma_bf16(a, b0, acc0);
      acc1 = wmma_bf16(a, b1f, acc1);
    }
  }

  const int c0 = cbase + col0, c1 = cbase + col0 + 64;
  const float bc0 = bias[c0], bc1 = bias[c1];
#pragma unroll
  for (int v = 0; v < 8; ++v) {
    const int row = row0 + v + half * 8;
    out[(size_t)row * 1024 + c0] = acc0[v] + bc0;
    out[(size_t)row * 1024 + c1] = acc1[v] + bc1;
  }
}

// ---------------------------------------------------------------------------
// Kernel 5: tiny 4-token attention per (batch, head).  Grid 256, 128 threads.
// ---------------------------------------------------------------------------
__global__ void attn_kernel(const float* __restrict__ q,
                            const float* __restrict__ k,
                            const float* __restrict__ v,
                            float* __restrict__ o) {
  const int b = blockIdx.x >> 3, h = blockIdx.x & 7;
  const int t = threadIdx.x;
  __shared__ float sq[4][128], sk[4][128], sv[4][128], sc[4][4], red[128];
#pragma unroll
  for (int l = 0; l < 4; ++l) {
    const size_t base = ((size_t)(b * 4 + l)) * COMMON + h * 128 + t;
    sq[l][t] = q[base]; sk[l][t] = k[base]; sv[l][t] = v[base];
  }
  __syncthreads();
  const float scale = 0.08838834764831845f;  // 1/sqrt(128)
  for (int l = 0; l < 4; ++l)
    for (int m = 0; m < 4; ++m) {
      red[t] = sq[l][t] * sk[m][t];
      __syncthreads();
      for (int off = 64; off > 0; off >>= 1) { if (t < off) red[t] += red[t + off]; __syncthreads(); }
      if (t == 0) sc[l][m] = red[0] * scale;
      __syncthreads();
    }
  if (t < 4) {
    float mx = fmaxf(fmaxf(sc[t][0], sc[t][1]), fmaxf(sc[t][2], sc[t][3]));
    float e0 = __expf(sc[t][0] - mx), e1 = __expf(sc[t][1] - mx);
    float e2 = __expf(sc[t][2] - mx), e3 = __expf(sc[t][3] - mx);
    float s = e0 + e1 + e2 + e3;
    sc[t][0] = e0 / s; sc[t][1] = e1 / s; sc[t][2] = e2 / s; sc[t][3] = e3 / s;
  }
  __syncthreads();
#pragma unroll
  for (int l = 0; l < 4; ++l) {
    float acc = sc[l][0] * sv[0][t] + sc[l][1] * sv[1][t] +
                sc[l][2] * sv[2][t] + sc[l][3] * sv[3][t];
    o[((size_t)(b * 4 + l)) * COMMON + h * 128 + t] = acc;
  }
}

// ---------------------------------------------------------------------------
// Kernel 6: expand + LayerNorm + AdaLN(time) -> normed[b,D]; time gate bias.
// ---------------------------------------------------------------------------
template <int D>
__global__ void ada_kernel(const float* __restrict__ y,
                           const float* __restrict__ expW,
                           const float* __restrict__ expb,
                           const float* __restrict__ tE,
                           const float* __restrict__ adaW,
                           const float* __restrict__ adab,
                           const float* __restrict__ gateW,
                           const float* __restrict__ gateb,
                           float* __restrict__ normed,
                           float* __restrict__ tbias, int lvl) {
  const int b = blockIdx.x, tid = threadIdx.x;
  __shared__ float yr[1024], tr[256], sExp[D], red[256];
  for (int k = tid; k < 1024; k += 256) yr[k] = y[((size_t)b * 4 + lvl) * COMMON + k];
  tr[tid] = tE[b * TIME_DIM + tid];
  __syncthreads();

  for (int c = tid; c < D; c += 256) {
    float acc = expb[c];
    for (int k = 0; k < 1024; ++k) acc += yr[k] * expW[(size_t)k * D + c];
    sExp[c] = acc;
  }
  __syncthreads();

  float s = 0.0f, s2 = 0.0f;
  for (int c = tid; c < D; c += 256) { float v = sExp[c]; s += v; s2 += v * v; }
  red[tid] = s; __syncthreads();
  for (int o = 128; o > 0; o >>= 1) { if (tid < o) red[tid] += red[tid + o]; __syncthreads(); }
  const float mu = red[0] / (float)D; __syncthreads();
  red[tid] = s2; __syncthreads();
  for (int o = 128; o > 0; o >>= 1) { if (tid < o) red[tid] += red[tid + o]; __syncthreads(); }
  const float var = red[0] / (float)D - mu * mu; __syncthreads();
  const float rstd = rsqrtf(var + 1e-5f);

  for (int c = tid; c < D; c += 256) {
    float s1 = adab[c], sB = adab[D + c];
    for (int k = 0; k < TIME_DIM; ++k) {
      const float tv = tr[k];
      s1 += tv * adaW[(size_t)k * (2 * D) + c];
      sB += tv * adaW[(size_t)k * (2 * D) + D + c];
    }
    const float ln = (sExp[c] - mu) * rstd;
    normed[(size_t)b * D + c] = ln * (1.0f + s1) + sB;
    float tb = gateb[c];
    for (int k = 0; k < TIME_DIM; ++k) tb += tr[k] * gateW[(size_t)(D + k) * D + c];
    tbias[(size_t)b * D + c] = tb;
  }
}

// ---------------------------------------------------------------------------
// Kernel 7: gate GEMM + fused sigmoid gate + residual output.
// out[row,c] = e[row,c] + sigmoid(e_row @ gateW[:D] + tbias[b]) * normed[b]
// Grid (B*S/16, D/128), 128 threads.
// ---------------------------------------------------------------------------
template <int D, int S>
__global__ void gate_kernel(const float* __restrict__ e,
                            const float* __restrict__ gateW,
                            const float* __restrict__ normed,
                            const float* __restrict__ tbias,
                            float* __restrict__ out) {
  __shared__ __bf16 lA[16 * D];
  __shared__ __bf16 lBT[128 * 64];
  const int tid = threadIdx.x, lane = tid & 31, wave = tid >> 5;
  const int half = lane >> 4, r = lane & 15;
  const int col0 = wave * 16 + r;
  const int row0 = blockIdx.x * 16, cbase = blockIdx.y * 128;
  const int b = row0 / S;

  stage_A_bf16(lA, e + (size_t)row0 * D, 16, D, D, tid, 128);
  __syncthreads();

  f32x8 acc0 = {0, 0, 0, 0, 0, 0, 0, 0};
  f32x8 acc1 = {0, 0, 0, 0, 0, 0, 0, 0};
  for (int k0 = 0; k0 < D; k0 += 64) {
    __syncthreads();
    stage_Bt_bf16(lBT, gateW + (size_t)k0 * D + cbase, D, tid);
    if (k0 + 64 < D)
      __builtin_prefetch(gateW + (size_t)(k0 + 64) * D + cbase, 0, 1);
    __syncthreads();
#pragma unroll
    for (int ks = 0; ks < 64; ks += 32) {
      bf16x16 a  = load_fragA(lA, D, k0 + ks, r, half);
      bf16x16 b0 = load_fragB(lBT, col0, ks, half);
      bf16x16 b1f = load_fragB(lBT, col0 + 64, ks, half);
      acc0 = wmma_bf16(a, b0, acc0);
      acc1 = wmma_bf16(a, b1f, acc1);
    }
  }

  const int c0 = cbase + col0, c1 = cbase + col0 + 64;
  const float tb0 = tbias[(size_t)b * D + c0], nm0 = normed[(size_t)b * D + c0];
  const float tb1 = tbias[(size_t)b * D + c1], nm1 = normed[(size_t)b * D + c1];
#pragma unroll
  for (int v = 0; v < 8; ++v) {
    const int row = row0 + v + half * 8;
    const size_t i0 = (size_t)row * D + c0;
    const size_t i1 = (size_t)row * D + c1;
    const float sg0 = 1.0f / (1.0f + __expf(-(acc0[v] + tb0)));
    const float sg1 = 1.0f / (1.0f + __expf(-(acc1[v] + tb1)));
    out[i0] = e[i0] + sg0 * nm0;
    out[i1] = e[i1] + sg1 * nm1;
  }
}

// ---------------------------------------------------------------------------
// Host launch.  Input ordering assumes JAX pytree flattening (dicts by sorted
// key): emb0..emb3, time_embed, levels[0..3]{adaW, adab, aggW1, aggW2, aggb1,
// aggb2, expW, expb, gateW, gateb, projW, projb}, mha{Wk, Wo, Wq, Wv, bk, bo,
// bq, bv}, pos.
// ---------------------------------------------------------------------------
enum { P_adaW = 0, P_adab, P_aggW1, P_aggW2, P_aggb1, P_aggb2,
       P_expW, P_expb, P_gateW, P_gateb, P_projW, P_projb };

extern "C" void kernel_launch(void* const* d_in, const int* in_sizes, int n_in,
                              void* d_out, int out_size, void* d_ws, size_t ws_size,
                              hipStream_t stream) {
  (void)in_sizes; (void)n_in; (void)out_size; (void)ws_size;
  const int DD[4] = {256, 512, 768, 1024};
  const int SS[4] = {2048, 1024, 512, 256};

  const float* emb[4];
  for (int i = 0; i < 4; ++i) emb[i] = (const float*)d_in[i];
  const float* tE = (const float*)d_in[4];
  auto LP = [&](int l, int j) { return (const float*)d_in[5 + l * 12 + j]; };
  const int mb = 5 + 4 * 12;
  const float* Wk = (const float*)d_in[mb + 0];
  const float* Wo = (const float*)d_in[mb + 1];
  const float* Wq = (const float*)d_in[mb + 2];
  const float* Wv = (const float*)d_in[mb + 3];
  const float* bk = (const float*)d_in[mb + 4];
  const float* bo = (const float*)d_in[mb + 5];
  const float* bq = (const float*)d_in[mb + 6];
  const float* bv = (const float*)d_in[mb + 7];
  const float* pos = (const float*)d_in[mb + 8];

  float* ws = (float*)d_ws;
  size_t off = 0;
  auto alloc = [&](size_t n) { float* p = ws + off; off += n; return p; };
  float* sc[4];     for (int l = 0; l < 4; ++l) sc[l]     = alloc((size_t)BATCH * SS[l]);
  float* pooled[4]; for (int l = 0; l < 4; ++l) pooled[l] = alloc((size_t)BATCH * DD[l]);
  float* x  = alloc(128 * 1024);
  float* qb = alloc(128 * 1024);
  float* kb = alloc(128 * 1024);
  float* vb = alloc(128 * 1024);
  float* ob = alloc(128 * 1024);
  float* yb = alloc(128 * 1024);
  float* normed[4]; for (int l = 0; l < 4; ++l) normed[l] = alloc((size_t)BATCH * DD[l]);
  float* tbias[4];  for (int l = 0; l < 4; ++l) tbias[l]  = alloc((size_t)BATCH * DD[l]);

  float* out = (float*)d_out;
  size_t outoff[4];
  { size_t o = 0; for (int l = 0; l < 4; ++l) { outoff[l] = o; o += (size_t)BATCH * SS[l] * DD[l]; } }

  // ---- stage 1: per-level scores + pooling + projection ----
  score_kernel<256, 2048><<<BATCH * 2048 / 16, 128, 0, stream>>>(emb[0], LP(0, P_aggW1), LP(0, P_aggb1), LP(0, P_aggW2), sc[0]);
  score_kernel<512, 1024><<<BATCH * 1024 / 16, 128, 0, stream>>>(emb[1], LP(1, P_aggW1), LP(1, P_aggb1), LP(1, P_aggW2), sc[1]);
  score_kernel<768, 512 ><<<BATCH * 512  / 16, 128, 0, stream>>>(emb[2], LP(2, P_aggW1), LP(2, P_aggb1), LP(2, P_aggW2), sc[2]);
  score_kernel<1024, 256><<<BATCH * 256  / 16, 128, 0, stream>>>(emb[3], LP(3, P_aggW1), LP(3, P_aggb1), LP(3, P_aggW2), sc[3]);

  pool_kernel<256, 2048><<<BATCH, 256, 0, stream>>>(emb[0], sc[0], pooled[0]);
  pool_kernel<512, 1024><<<BATCH, 256, 0, stream>>>(emb[1], sc[1], pooled[1]);
  pool_kernel<768, 512 ><<<BATCH, 256, 0, stream>>>(emb[2], sc[2], pooled[2]);
  pool_kernel<1024, 256><<<BATCH, 256, 0, stream>>>(emb[3], sc[3], pooled[3]);

  for (int l = 0; l < 4; ++l)
    proj_kernel<<<BATCH, 256, 0, stream>>>(pooled[l], LP(l, P_projW), LP(l, P_projb), pos, x, DD[l], l);

  // ---- stage 2: MHA over the 4 level tokens ----
  dim3 lg(8, 8);
  linear128_kernel<<<lg, 128, 0, stream>>>(x, Wq, bq, qb);
  linear128_kernel<<<lg, 128, 0, stream>>>(x, Wk, bk, kb);
  linear128_kernel<<<lg, 128, 0, stream>>>(x, Wv, bv, vb);
  attn_kernel<<<BATCH * 8, 128, 0, stream>>>(qb, kb, vb, ob);
  linear128_kernel<<<lg, 128, 0, stream>>>(ob, Wo, bo, yb);

  // ---- stage 3: expand + AdaLN + time gate bias ----
  ada_kernel<256 ><<<BATCH, 256, 0, stream>>>(yb, LP(0, P_expW), LP(0, P_expb), tE, LP(0, P_adaW), LP(0, P_adab), LP(0, P_gateW), LP(0, P_gateb), normed[0], tbias[0], 0);
  ada_kernel<512 ><<<BATCH, 256, 0, stream>>>(yb, LP(1, P_expW), LP(1, P_expb), tE, LP(1, P_adaW), LP(1, P_adab), LP(1, P_gateW), LP(1, P_gateb), normed[1], tbias[1], 1);
  ada_kernel<768 ><<<BATCH, 256, 0, stream>>>(yb, LP(2, P_expW), LP(2, P_expb), tE, LP(2, P_adaW), LP(2, P_adab), LP(2, P_gateW), LP(2, P_gateb), normed[2], tbias[2], 2);
  ada_kernel<1024><<<BATCH, 256, 0, stream>>>(yb, LP(3, P_expW), LP(3, P_expb), tE, LP(3, P_adaW), LP(3, P_adab), LP(3, P_gateW), LP(3, P_gateb), normed[3], tbias[3], 3);

  // ---- stage 4: fused gate GEMM + sigmoid + residual write ----
  gate_kernel<256, 2048><<<dim3(BATCH * 2048 / 16, 256 / 128),  128, 0, stream>>>(emb[0], LP(0, P_gateW), normed[0], tbias[0], out + outoff[0]);
  gate_kernel<512, 1024><<<dim3(BATCH * 1024 / 16, 512 / 128),  128, 0, stream>>>(emb[1], LP(1, P_gateW), normed[1], tbias[1], out + outoff[1]);
  gate_kernel<768, 512 ><<<dim3(BATCH * 512  / 16, 768 / 128),  128, 0, stream>>>(emb[2], LP(2, P_gateW), normed[2], tbias[2], out + outoff[2]);
  gate_kernel<1024, 256><<<dim3(BATCH * 256  / 16, 1024 / 128), 128, 0, stream>>>(emb[3], LP(3, P_gateW), normed[3], tbias[3], out + outoff[3]);
}
